// ModalityGatedMILBinaryHead_35012573397301
// MI455X (gfx1250) — matrix operand order repaired and checked by reading, not compiled, not run on hardware
//
#include <hip/hip_runtime.h>
#include <hip/hip_bf16.h>
#include <math.h>

#define D_    768
#define H_    256
#define NTOK  4096
#define NP1   4097
#define MMOD  4
#define KKS   24            // 768 / 32
#define ROWS_BF 784         // padded bf16 row stride in LDS
#define GROUP_BYTES (16 * ROWS_BF * 2)   // 25088 B per 16-token group

typedef __attribute__((ext_vector_type(16))) __bf16 v16bf;
typedef __attribute__((ext_vector_type(8)))  float  v8f;

union ABfrag { uint4 q[2]; v16bf v; };

static __device__ __forceinline__ unsigned short f2bf(float f) {
    unsigned u = __float_as_uint(f);
    unsigned r = (u + 0x7FFFu + ((u >> 16) & 1u)) >> 16;
    return (unsigned short)r;
}

// ---------------------------------------------------------------------------
// Kernel 1: repack attn_w1 (768x256 fp32, row-major over D) into bf16 WMMA
// B-fragment layout: element index = ((kk*16 + t)*32 + lane)*16 + i
//   lanes 0-15:  K = kk*32 + i      (i=0..15), N = lane
//   lanes 16-31: K = kk*32 + 16 + i,           N = lane-16
// Each lane's 16 bf16 (32 bytes) are contiguous -> two b128 loads in the GEMM.
// ---------------------------------------------------------------------------
__global__ __launch_bounds__(256) void wprep_kernel(const float* __restrict__ w1,
                                                    unsigned short* __restrict__ wfrag) {
    int idx = blockIdx.x * 256 + threadIdx.x;           // 768*256 elements exactly
    int i    = idx & 15;
    int lane = (idx >> 4) & 31;
    int t    = (idx >> 9) & 15;
    int kk   = idx >> 13;
    int d = kk * 32 + ((lane >> 4) << 4) + i;
    int h = (t << 4) + (lane & 15);
    wfrag[idx] = f2bf(w1[d * H_ + h]);
}

// ---------------------------------------------------------------------------
// Kernel 2: fused LN -> bf16 -> WMMA GEMM (D=768 x H=256) -> tanh -> dot w2
// Block = 256 threads (8 waves), 64 tokens per block (4 groups of 16).
// Wave pair (2w, 2w+1) shares one 16-token group; each wave computes half of H.
// ---------------------------------------------------------------------------
__global__ __launch_bounds__(256) void attn_score_kernel(
    const float* __restrict__ fused,
    const float* __restrict__ ln_g, const float* __restrict__ ln_b,
    const unsigned short* __restrict__ wfrag,
    const float* __restrict__ b1, const float* __restrict__ w2,
    const float* __restrict__ b2,
    float* __restrict__ scores)
{
    extern __shared__ char smem[];                       // 4*25088 + 256 bytes
    float* sScore = (float*)(smem + 4 * GROUP_BYTES);

    const int tid  = threadIdx.x;
    const int lane = tid & 31;
    const int wave = tid >> 5;
    const int b    = blockIdx.y;
    const int tileBase = blockIdx.x * 64;

    if (tid < 64) sScore[tid] = 0.0f;

    // ---------------- LayerNorm + bf16 pack into LDS ----------------
    const int g     = wave >> 1;       // token group 0..3
    const int rhalf = wave & 1;        // which 8 rows of the group
    unsigned short* aGroup = (unsigned short*)(smem + g * GROUP_BYTES);
    const float* tokBase = fused +
        ((size_t)b * NP1 + 1 + tileBase + g * 16 + rhalf * 8) * D_;

    for (int r = 0; r < 8; ++r) {
        const float4* row = (const float4*)(tokBase + (size_t)r * D_);
        float x[6][4];
        float s = 0.f, sq = 0.f;
        #pragma unroll
        for (int k = 0; k < 6; ++k) {
            float4 v = row[lane + k * 32];
            x[k][0] = v.x; x[k][1] = v.y; x[k][2] = v.z; x[k][3] = v.w;
            s  += v.x + v.y + v.z + v.w;
            sq += v.x * v.x + v.y * v.y + v.z * v.z + v.w * v.w;
        }
        #pragma unroll
        for (int m = 16; m >= 1; m >>= 1) {
            s  += __shfl_xor(s,  m, 32);
            sq += __shfl_xor(sq, m, 32);
        }
        float mean = s * (1.0f / 768.0f);
        float var  = sq * (1.0f / 768.0f) - mean * mean;
        float ri   = rsqrtf(var + 1e-5f);
        unsigned short* outRow = aGroup + (rhalf * 8 + r) * ROWS_BF;
        #pragma unroll
        for (int k = 0; k < 6; ++k) {
            int d4 = lane + k * 32;
            float4 gg = ((const float4*)ln_g)[d4];
            float4 bb = ((const float4*)ln_b)[d4];
            float y0 = (x[k][0] - mean) * ri * gg.x + bb.x;
            float y1 = (x[k][1] - mean) * ri * gg.y + bb.y;
            float y2 = (x[k][2] - mean) * ri * gg.z + bb.z;
            float y3 = (x[k][3] - mean) * ri * gg.w + bb.w;
            uint2 p;
            p.x = (unsigned)f2bf(y0) | ((unsigned)f2bf(y1) << 16);
            p.y = (unsigned)f2bf(y2) | ((unsigned)f2bf(y3) << 16);
            *(uint2*)(outRow + d4 * 4) = p;
        }
    }
    __syncthreads();

    // ---------------- WMMA GEMM: 16 tokens x 128 H-cols per wave ----------------
    const int hh   = wave & 1;                 // H half: cols hh*128 .. +128
    const char* aBase = smem + g * GROUP_BYTES;
    const int arow  = lane & 15;
    const int akoff = (lane >> 4) * 16;        // A-frag K offset (elements)
    const uint4* wq = (const uint4*)wfrag;

    v8f acc[8];
    v8f zero = {0.f, 0.f, 0.f, 0.f, 0.f, 0.f, 0.f, 0.f};
    #pragma unroll
    for (int t = 0; t < 8; ++t) acc[t] = zero;

    for (int kk = 0; kk < KKS; ++kk) {
        ABfrag a;
        int aoff = arow * (ROWS_BF * 2) + (kk * 32 + akoff) * 2;
        a.q[0] = *(const uint4*)(aBase + aoff);
        a.q[1] = *(const uint4*)(aBase + aoff + 16);
        #pragma unroll
        for (int t = 0; t < 8; ++t) {
            size_t off = ((size_t)((kk * 16 + hh * 8 + t) * 32) + lane) * 2;
            ABfrag bf;
            bf.q[0] = wq[off];
            bf.q[1] = wq[off + 1];
            acc[t] = __builtin_amdgcn_wmma_f32_16x16x32_bf16(
                false, a.v, false, bf.v, (short)0, acc[t], false, false);
        }
    }

    // ---------------- epilogue: scores = sum_h tanh(h + b1) * w2 ----------------
    float scv[8];
    #pragma unroll
    for (int v = 0; v < 8; ++v) scv[v] = 0.f;
    #pragma unroll
    for (int t = 0; t < 8; ++t) {
        int col = hh * 128 + t * 16 + (lane & 15);
        float w2c = w2[col];
        float b1c = b1[col];
        #pragma unroll
        for (int v = 0; v < 8; ++v)
            scv[v] += tanhf(acc[t][v] + b1c) * w2c;
    }
    #pragma unroll
    for (int v = 0; v < 8; ++v) {
        #pragma unroll
        for (int m = 1; m < 16; m <<= 1)
            scv[v] += __shfl_xor(scv[v], m, 32);
    }
    if ((lane & 15) == 0) {
        #pragma unroll
        for (int v = 0; v < 8; ++v) {
            int tok = g * 16 + (lane >> 4) * 8 + v;   // C layout: lanes>=16 hold M=v+8
            atomicAdd(&sScore[tok], scv[v]);
        }
    }
    __syncthreads();
    if (tid < 64)
        scores[(size_t)b * NTOK + tileBase + tid] = sScore[tid] + b2[0];
}

// ---------------------------------------------------------------------------
// Kernel 3: per-(b,m) masked softmax stats: max, sum(exp), valid
// ---------------------------------------------------------------------------
__global__ __launch_bounds__(256) void stats_kernel(
    const float* __restrict__ scores,
    const unsigned char* __restrict__ token_mask,
    const int* __restrict__ modality_ids,
    float* __restrict__ mx, float* __restrict__ den, float* __restrict__ valid)
{
    __shared__ float red[256], red2[256];
    int tid = threadIdx.x, m = blockIdx.x, b = blockIdx.y;
    float sv[16]; unsigned mk = 0;
    float lmax = -INFINITY, lcnt = 0.f;
    #pragma unroll
    for (int i = 0; i < 16; ++i) {
        int n = tid + i * 256;
        bool ok = (token_mask[n] != 0) && (modality_ids[n] == m);
        float s = scores[(size_t)b * NTOK + n];
        sv[i] = s;
        if (ok) { mk |= (1u << i); lmax = fmaxf(lmax, s); lcnt += 1.f; }
    }
    red[tid] = lmax; red2[tid] = lcnt; __syncthreads();
    for (int s = 128; s > 0; s >>= 1) {
        if (tid < s) { red[tid] = fmaxf(red[tid], red[tid + s]); red2[tid] += red2[tid + s]; }
        __syncthreads();
    }
    float gmax = red[0], gcnt = red2[0];
    __syncthreads();
    float lsum = 0.f;
    #pragma unroll
    for (int i = 0; i < 16; ++i)
        if (mk & (1u << i)) lsum += expf(sv[i] - gmax);
    red[tid] = lsum; __syncthreads();
    for (int s = 128; s > 0; s >>= 1) {
        if (tid < s) red[tid] += red[tid + s];
        __syncthreads();
    }
    if (tid == 0) {
        int idx = b * MMOD + m;
        mx[idx]    = (gcnt > 0.f) ? gmax : 0.f;
        den[idx]   = fmaxf(red[0], 1.17549435e-38f);
        valid[idx] = (gcnt > 0.f) ? 1.f : 0.f;
    }
}

__global__ __launch_bounds__(256) void zero_pools_kernel(float* __restrict__ pools) {
    int i = blockIdx.x * 256 + threadIdx.x;   // 32*4*768 = 98304 exactly
    pools[i] = 0.f;
}

// ---------------------------------------------------------------------------
// Kernel 4: softmax weights + token_attention + weighted pooling (streaming)
// grid (seg=16, b=32): each block handles 256 tokens, accumulates partial
// pools in registers, merges with global fp32 atomics.
// ---------------------------------------------------------------------------
__global__ __launch_bounds__(256) void pool_kernel(
    const float* __restrict__ fused, const float* __restrict__ scores,
    const unsigned char* __restrict__ token_mask, const int* __restrict__ modality_ids,
    const float* __restrict__ mx, const float* __restrict__ den,
    float* __restrict__ pools, float* __restrict__ token_attn)
{
    __shared__ float shw[256 * 4];
    int tid = threadIdx.x, seg = blockIdx.x, b = blockIdx.y;
    int n = seg * 256 + tid;
    int m = modality_ids[n];
    float w = 0.f;
    if (token_mask[n]) {
        int idx = b * MMOD + m;
        w = expf(scores[(size_t)b * NTOK + n] - mx[idx]) / den[idx];
    }
    token_attn[(size_t)b * NTOK + n] = w;
    #pragma unroll
    for (int mm = 0; mm < 4; ++mm) shw[tid * 4 + mm] = (mm == m) ? w : 0.f;
    __syncthreads();

    const float* tokSeg = fused + ((size_t)b * NP1 + 1 + seg * 256) * D_;
    float a0[3] = {0,0,0}, a1[3] = {0,0,0}, a2[3] = {0,0,0}, a3[3] = {0,0,0};
    for (int j = 0; j < 256; ++j) {
        float4 wv = *(const float4*)&shw[j * 4];
        const float* row = tokSeg + (size_t)j * D_;
        #pragma unroll
        for (int c = 0; c < 3; ++c) {
            float x = row[tid + c * 256];
            a0[c] += wv.x * x; a1[c] += wv.y * x;
            a2[c] += wv.z * x; a3[c] += wv.w * x;
        }
    }
    #pragma unroll
    for (int c = 0; c < 3; ++c) {
        int d = tid + c * 256;
        atomicAdd(&pools[((size_t)b * 4 + 0) * D_ + d], a0[c]);
        atomicAdd(&pools[((size_t)b * 4 + 1) * D_ + d], a1[c]);
        atomicAdd(&pools[((size_t)b * 4 + 2) * D_ + d], a2[c]);
        atomicAdd(&pools[((size_t)b * 4 + 3) * D_ + d], a3[c]);
    }
}

// ---------------------------------------------------------------------------
// Kernel 5: expert + gate heads, gate softmax, final logits (tiny)
// ---------------------------------------------------------------------------
__global__ __launch_bounds__(256) void head_kernel(
    const float* __restrict__ fused, const float* __restrict__ pools,
    const float* __restrict__ valid,
    const float* __restrict__ exp_g,  const float* __restrict__ exp_b,
    const float* __restrict__ exp_w1, const float* __restrict__ exp_b1,
    const float* __restrict__ exp_w2, const float* __restrict__ exp_b2,
    const float* __restrict__ gate_g,  const float* __restrict__ gate_b,
    const float* __restrict__ gate_w1, const float* __restrict__ gate_b1,
    const float* __restrict__ gate_w2, const float* __restrict__ gate_b2,
    const float* __restrict__ cls_w,   const float* __restrict__ cls_b,
    float* __restrict__ out_logits, float* __restrict__ out_gw,
    float* __restrict__ out_el)
{
    __shared__ float sp[4 * 768];
    __shared__ float scl[768];
    __shared__ float red[256], red2[256];
    __shared__ float emean[4], erinv[4], psum[4], psq[4];
    __shared__ float elog[4], gsc[4];
    __shared__ float csum_s, csq_s;
    int tid = threadIdx.x, b = blockIdx.x;

    for (int i = tid; i < 3072; i += 256) sp[i] = pools[(size_t)b * 3072 + i];
    for (int i = tid; i < 768;  i += 256) scl[i] = fused[(size_t)b * NP1 * D_ + i];
    __syncthreads();

    float ls = 0.f, lq = 0.f;
    #pragma unroll
    for (int c = 0; c < 3; ++c) { float v = scl[tid + c * 256]; ls += v; lq += v * v; }
    red[tid] = ls; red2[tid] = lq; __syncthreads();
    for (int s = 128; s > 0; s >>= 1) {
        if (tid < s) { red[tid] += red[tid + s]; red2[tid] += red2[tid + s]; }
        __syncthreads();
    }
    if (tid == 0) { csum_s = red[0]; csq_s = red2[0]; }
    __syncthreads();

    for (int m = 0; m < 4; ++m) {
        float ps = 0.f, pq = 0.f;
        #pragma unroll
        for (int c = 0; c < 3; ++c) { float v = sp[m * 768 + tid + c * 256]; ps += v; pq += v * v; }
        red[tid] = ps; red2[tid] = pq; __syncthreads();
        for (int s = 128; s > 0; s >>= 1) {
            if (tid < s) { red[tid] += red[tid + s]; red2[tid] += red2[tid + s]; }
            __syncthreads();
        }
        if (tid == 0) {
            float mean = red[0] * (1.f / 768.f);
            float var  = red2[0] * (1.f / 768.f) - mean * mean;
            emean[m] = mean; erinv[m] = rsqrtf(var + 1e-5f);
            psum[m] = red[0]; psq[m] = red2[0];
        }
        __syncthreads();
    }

    // expert logits
    for (int m = 0; m < 4; ++m) {
        float mean = emean[m], ri = erinv[m];
        float acc = 0.f;
        for (int d = 0; d < 768; ++d) {
            float xln = (sp[m * 768 + d] - mean) * ri * exp_g[d] + exp_b[d];
            acc += xln * exp_w1[d * 256 + tid];
        }
        float h = acc + exp_b1[tid];
        float gel = 0.5f * h * (1.f + erff(h * 0.70710678118f));
        red[tid] = gel * exp_w2[tid]; __syncthreads();
        for (int s = 128; s > 0; s >>= 1) { if (tid < s) red[tid] += red[tid + s]; __syncthreads(); }
        if (tid == 0) elog[m] = red[0] + exp_b2[0];
        __syncthreads();
    }

    // gate scores
    for (int m = 0; m < 4; ++m) {
        float mean2 = (csum_s + psum[m]) * (1.f / 1536.f);
        float var2  = (csq_s + psq[m]) * (1.f / 1536.f) - mean2 * mean2;
        float ri2   = rsqrtf(var2 + 1e-5f);
        float acc = 0.f;
        for (int d = 0; d < 768; ++d) {
            float xln = (scl[d] - mean2) * ri2 * gate_g[d] + gate_b[d];
            acc += xln * gate_w1[d * 256 + tid];
        }
        for (int d = 0; d < 768; ++d) {
            float xln = (sp[m * 768 + d] - mean2) * ri2 * gate_g[768 + d] + gate_b[768 + d];
            acc += xln * gate_w1[(768 + d) * 256 + tid];
        }
        float h = acc + gate_b1[tid];
        float gel = 0.5f * h * (1.f + erff(h * 0.70710678118f));
        red[tid] = gel * gate_w2[tid]; __syncthreads();
        for (int s = 128; s > 0; s >>= 1) { if (tid < s) red[tid] += red[tid + s]; __syncthreads(); }
        if (tid == 0) gsc[m] = red[0] + gate_b2[0];
        __syncthreads();
    }

    // cls logit + gate softmax + outputs
    float cp = 0.f;
    #pragma unroll
    for (int c = 0; c < 3; ++c) { int d = tid + c * 256; cp += scl[d] * cls_w[d]; }
    red[tid] = cp; __syncthreads();
    for (int s = 128; s > 0; s >>= 1) { if (tid < s) red[tid] += red[tid + s]; __syncthreads(); }
    if (tid == 0) {
        float mxg = -INFINITY; bool any = false;
        for (int m = 0; m < 4; ++m)
            if (valid[b * 4 + m] > 0.5f) { any = true; mxg = fmaxf(mxg, gsc[m]); }
        if (!any) mxg = 0.f;
        float e[4], s = 0.f;
        for (int m = 0; m < 4; ++m) {
            e[m] = (valid[b * 4 + m] > 0.5f) ? expf(gsc[m] - mxg) : 0.f;
            s += e[m];
        }
        float dn = fmaxf(s, 1.17549435e-38f);
        float mixed = 0.f;
        for (int m = 0; m < 4; ++m) {
            float gw = e[m] / dn;
            out_gw[b * 4 + m] = gw;
            out_el[b * 4 + m] = elog[m];
            mixed += gw * elog[m];
        }
        out_logits[b] = red[0] + cls_b[0] + mixed;
    }
}

// ---------------------------------------------------------------------------
extern "C" void kernel_launch(void* const* d_in, const int* in_sizes, int n_in,
                              void* d_out, int out_size, void* d_ws, size_t ws_size,
                              hipStream_t stream) {
    const float*         fused        = (const float*)d_in[0];
    const unsigned char* token_mask   = (const unsigned char*)d_in[1];
    const int*           modality_ids = (const int*)d_in[2];
    // d_in[3] = num_modalities (== 4, hardcoded for this problem instance)
    const float* attn_ln_g = (const float*)d_in[4];
    const float* attn_ln_b = (const float*)d_in[5];
    const float* attn_w1   = (const float*)d_in[6];
    const float* attn_b1   = (const float*)d_in[7];
    const float* attn_w2   = (const float*)d_in[8];
    const float* attn_b2   = (const float*)d_in[9];
    const float* exp_ln_g  = (const float*)d_in[10];
    const float* exp_ln_b  = (const float*)d_in[11];
    const float* exp_w1    = (const float*)d_in[12];
    const float* exp_b1    = (const float*)d_in[13];
    const float* exp_w2    = (const float*)d_in[14];
    const float* exp_b2    = (const float*)d_in[15];
    const float* gate_ln_g = (const float*)d_in[16];
    const float* gate_ln_b = (const float*)d_in[17];
    const float* gate_w1   = (const float*)d_in[18];
    const float* gate_b1   = (const float*)d_in[19];
    const float* gate_w2   = (const float*)d_in[20];
    const float* gate_b2   = (const float*)d_in[21];
    const float* cls_w     = (const float*)d_in[22];
    const float* cls_b     = (const float*)d_in[23];

    // workspace carve-up (~1.31 MB)
    char* ws = (char*)d_ws;
    unsigned short* wfrag = (unsigned short*)ws;                 // 768*256*2 = 393216
    float* scores = (float*)(ws + 393216);                       // 32*4096*4 = 524288
    float* mxv    = scores + 32 * NTOK;                          // 128
    float* den    = mxv + 128;                                   // 128
    float* valid  = den + 128;                                   // 128
    float* pools  = valid + 128;                                 // 32*4*768

    // output layout: logits[32] | gate_weights[128] | expert_logits[128] | token_attention[131072]
    float* out        = (float*)d_out;
    float* out_logits = out;
    float* out_gw     = out + 32;
    float* out_el     = out + 160;
    float* out_ta     = out + 288;

    wprep_kernel<<<768, 256, 0, stream>>>(attn_w1, wfrag);

    size_t smem = 4 * GROUP_BYTES + 64 * sizeof(float);          // 100608 B (<320KB WGP LDS)
    attn_score_kernel<<<dim3(NTOK / 64, 32), 256, smem, stream>>>(
        fused, attn_ln_g, attn_ln_b, wfrag, attn_b1, attn_w2, attn_b2, scores);

    stats_kernel<<<dim3(MMOD, 32), 256, 0, stream>>>(
        scores, token_mask, modality_ids, mxv, den, valid);

    zero_pools_kernel<<<384, 256, 0, stream>>>(pools);

    pool_kernel<<<dim3(16, 32), 256, 0, stream>>>(
        fused, scores, token_mask, modality_ids, mxv, den, pools, out_ta);

    head_kernel<<<32, 256, 0, stream>>>(
        fused, pools, valid,
        exp_ln_g, exp_ln_b, exp_w1, exp_b1, exp_w2, exp_b2,
        gate_ln_g, gate_ln_b, gate_w1, gate_b1, gate_w2, gate_b2,
        cls_w, cls_b, out_logits, out_gw, out_el);
}
